// Attention_87625922773715
// MI455X (gfx1250) — matrix operand order repaired and checked by reading, compiled, not run on hardware
//
#include <hip/hip_runtime.h>
#include <hip/hip_bf16.h>

// ---------------------------------------------------------------------------
// SAGAN-style attention block for MI455X (gfx1250), bf16 WMMA everywhere.
// B=8, C=256, H=W=64 (n=4096), HEADS=4, pooled m=1024, d_qk=64, d_v=32.
// GEMMs use async global->LDS tile staging (ASYNCcnt) with double buffering.
// ---------------------------------------------------------------------------

typedef unsigned int  uint_t;
typedef unsigned short ushort_t;

typedef __attribute__((ext_vector_type(16))) __bf16 bf16x16;
typedef __attribute__((ext_vector_type(8)))  float  f32x8;

union FragB {
    bf16x16 v;
    uint4   q[2];
};

static __device__ __forceinline__ ushort_t f2bf(float f) {
    union { float f; uint_t u; } x; x.f = f;
    uint_t r = (x.u + 0x7FFFu + ((x.u >> 16) & 1u)) >> 16;
    return (ushort_t)r;
}

static __device__ __forceinline__ f32x8 zero8() {
    f32x8 z;
#pragma unroll
    for (int i = 0; i < 8; ++i) z[i] = 0.0f;
    return z;
}

// LDS byte offset from a generic pointer to a __shared__ object: on AMDGPU the
// shared aperture occupies the high 32 bits, so addr[31:0] is the DS offset.
static __device__ __forceinline__ unsigned lds_off(const void* p) {
    return (unsigned)(unsigned long long)p;
}

// Async global->LDS copy of 16 bytes (per lane), tracked by ASYNCcnt.
static __device__ __forceinline__ void async_copy_b128(unsigned lds_addr, const void* gaddr) {
    asm volatile("global_load_async_to_lds_b128 %0, %1, off"
                 :: "v"(lds_addr), "v"(gaddr)
                 : "memory");
}

static __device__ __forceinline__ void wait_asynccnt0() {
    asm volatile("s_wait_asynccnt 0x0" ::: "memory");
}

// ---------------------------------------------------------------------------
// Packing kernels (fp32 -> bf16)
// ---------------------------------------------------------------------------

__global__ void pack_bf16(const float* __restrict__ s, ushort_t* __restrict__ d, int n) {
    int i = blockIdx.x * 256 + threadIdx.x;
    if (i < n) d[i] = f2bf(s[i]);
}

// x [b, c(256), n(4096)] f32  ->  xT [b*n, 256] bf16 (K contiguous for B-frags)
__global__ void pack_xT(const float* __restrict__ x, ushort_t* __restrict__ xT) {
    size_t idx = (size_t)blockIdx.x * 256 + threadIdx.x;   // B*C*N = 8*256*4096
    int i = (int)(idx & 4095);
    int c = (int)((idx >> 12) & 255);
    int b = (int)(idx >> 20);
    xT[(((size_t)b << 12) + i) * 256 + c] = f2bf(x[idx]);
}

// ---------------------------------------------------------------------------
// Generic bf16 GEMM: Y[m,p] = sum_k A[m,k] * X[k,p], X given as xT[p][k].
// Block tile 128(M) x 128(N), 8 waves (2x4), wave tile 64x32, k-step 32.
// LDS tiles are filled with GLOBAL_LOAD_ASYNC_TO_LDS_B128, double buffered.
// MODE 0: Y f32 -> out[b][m][i] (Mo channels).  MODE 1: out = gamma*Y + xres.
// ---------------------------------------------------------------------------

template <int MODE>
__global__ __launch_bounds__(256) void gemm_bf16_kernel(
    const ushort_t* __restrict__ A,    // [Mo, K] bf16 row-major
    const ushort_t* __restrict__ Bm,   // [Npix, K] bf16 row-major (transposed X)
    float* __restrict__ Y,
    const float* __restrict__ Xres,
    const float* __restrict__ gamma,
    int Mo, int K)
{
    __shared__ __align__(16) ushort_t As[2][128 * 32];
    __shared__ __align__(16) ushort_t Bs[2][128 * 32];

    const int tid  = threadIdx.x;
    const int wave = tid >> 5;
    const int lane = tid & 31;
    const int ln   = lane & 15;
    const int hl   = lane >> 4;
    const int wm   = wave & 1;        // 2 wave rows (M)
    const int wn   = wave >> 1;       // 4 wave cols (N)
    const int mtile = blockIdx.y * 128;
    const int ntile = blockIdx.x * 128;

    f32x8 acc[4][2];
#pragma unroll
    for (int fr = 0; fr < 4; ++fr)
#pragma unroll
        for (int fc = 0; fc < 2; ++fc) acc[fr][fc] = zero8();

    const int row = tid >> 1;         // 128 rows, 2 threads/row
    const int hb  = tid & 1;          // 32B half of the 64B row
    const ushort_t* ag = A  + (size_t)(mtile + row) * K + hb * 16;
    const ushort_t* bg = Bm + (size_t)(ntile + row) * K + hb * 16;
    const unsigned laA0 = lds_off(&As[0][row * 32 + hb * 16]);
    const unsigned laA1 = lds_off(&As[1][row * 32 + hb * 16]);
    const unsigned laB0 = lds_off(&Bs[0][row * 32 + hb * 16]);
    const unsigned laB1 = lds_off(&Bs[1][row * 32 + hb * 16]);

    // Issue one 32x(128+128) tile-pair of async copies (4x16B per thread).
    auto issue_tile = [&](int buf, int k0) {
        unsigned la = buf ? laA1 : laA0;
        unsigned lb = buf ? laB1 : laB0;
        async_copy_b128(la,       ag + k0);
        async_copy_b128(la + 16u, ag + k0 + 8);
        async_copy_b128(lb,       bg + k0);
        async_copy_b128(lb + 16u, bg + k0 + 8);
    };

    const int T = K >> 5;
    issue_tile(0, 0);

    for (int t = 0; t < T; ++t) {
        wait_asynccnt0();              // our async writes for tile t landed
        __syncthreads();               // ...and everyone else's too; also
                                       // orders next issue after all waves
                                       // finished reading buf[(t+1)&1] (t-1)
        if (t + 1 < T) issue_tile((t + 1) & 1, (t + 1) << 5);

        const ushort_t* Ab = As[t & 1];
        const ushort_t* Bb = Bs[t & 1];

        FragB Af[4], Bf[2];
#pragma unroll
        for (int fr = 0; fr < 4; ++fr) {
            int ra = wm * 64 + fr * 16 + ln;
            Af[fr].q[0] = *(const uint4*)&Ab[ra * 32 + hl * 8];        // K = hl*8 + 0..7
            Af[fr].q[1] = *(const uint4*)&Ab[ra * 32 + 16 + hl * 8];   // K = 16 + hl*8 + 0..7
        }
#pragma unroll
        for (int fc = 0; fc < 2; ++fc) {
            int rb = wn * 32 + fc * 16 + ln;
            Bf[fc].q[0] = *(const uint4*)&Bb[rb * 32 + hl * 8];        // K = hl*8 + 0..7
            Bf[fc].q[1] = *(const uint4*)&Bb[rb * 32 + 16 + hl * 8];   // K = 16 + hl*8 + 0..7
        }
#pragma unroll
        for (int fr = 0; fr < 4; ++fr)
#pragma unroll
            for (int fc = 0; fc < 2; ++fc)
                acc[fr][fc] = __builtin_amdgcn_wmma_f32_16x16x32_bf16(
                    false, Af[fr].v, false, Bf[fc].v, (short)0, acc[fr][fc], false, false);
    }

    const float gm = (MODE == 1) ? gamma[0] : 0.0f;
#pragma unroll
    for (int fr = 0; fr < 4; ++fr)
#pragma unroll
        for (int fc = 0; fc < 2; ++fc)
#pragma unroll
            for (int r = 0; r < 8; ++r) {
                int mo = mtile + wm * 64 + fr * 16 + r + hl * 8;
                int p  = ntile + wn * 32 + fc * 16 + ln;
                int b  = p >> 12;
                int i  = p & 4095;
                float v = acc[fr][fc][r];
                if (MODE == 0) {
                    Y[((size_t)b * Mo + mo) * 4096 + i] = v;
                } else {
                    size_t idx = ((size_t)b * 256 + mo) * 4096 + i;
                    Y[idx] = gm * v + Xres[idx];
                }
            }
}

// ---------------------------------------------------------------------------
// 2x2 max pool: in [b, Cc, 64, 64] -> out [b, Cc, 32, 32]
// ---------------------------------------------------------------------------

__global__ void pool2_kernel(const float* __restrict__ in, float* __restrict__ out, int Cc) {
    int idx = blockIdx.x * 256 + threadIdx.x;              // B*Cc*1024
    int j  = idx & 1023;
    int c  = (idx >> 10) % Cc;
    int b  = idx / (1024 * Cc);
    int jy = j >> 5, jx = j & 31;
    const float* p = in + (((size_t)(b * Cc + c) * 64 + jy * 2) * 64 + jx * 2);
    out[idx] = fmaxf(fmaxf(p[0], p[1]), fmaxf(p[64], p[65]));
}

// ---------------------------------------------------------------------------
// InstanceNorm over n=4096, output theta_n [b, h, i, ch(64)] bf16.
// channel c = ch*4 + h  (head is inner per reference reshape).
// ---------------------------------------------------------------------------

__global__ __launch_bounds__(256) void inorm_theta_kernel(
    const float* __restrict__ y, const float* __restrict__ w, const float* __restrict__ bias,
    ushort_t* __restrict__ outn)
{
    int bc = blockIdx.x;                 // B*256
    int c = bc & 255, b = bc >> 8;
    const float* p = y + (size_t)bc * 4096;

    float s = 0.f, ss = 0.f;
    for (int i = threadIdx.x; i < 4096; i += 256) { float v = p[i]; s += v; ss += v * v; }
    __shared__ float r1[256], r2[256];
    r1[threadIdx.x] = s; r2[threadIdx.x] = ss; __syncthreads();
    for (int st = 128; st > 0; st >>= 1) {
        if (threadIdx.x < st) { r1[threadIdx.x] += r1[threadIdx.x + st]; r2[threadIdx.x] += r2[threadIdx.x + st]; }
        __syncthreads();
    }
    float mean = r1[0] * (1.0f / 4096.0f);
    float var  = r2[0] * (1.0f / 4096.0f) - mean * mean;
    float rstd = rsqrtf(var + 1e-5f);
    float sw = w[c] * rstd;
    float sb = bias[c] - mean * sw;

    int head = c & 3, ch = c >> 2;
    ushort_t* o = outn + ((size_t)(b * 4 + head) * 4096) * 64 + ch;
    for (int i = threadIdx.x; i < 4096; i += 256)
        o[(size_t)i * 64] = f2bf(p[i] * sw + sb);
}

// ---------------------------------------------------------------------------
// InstanceNorm over m=1024 on pooled tensors.
// LAY 0 (phi, Cc=256): out [b, h, j, ch(64)]
// LAY 1 (g,   Cc=128): out [b, h, chv(32), j(1024)]
// ---------------------------------------------------------------------------

template <int LAY>
__global__ __launch_bounds__(256) void inorm_pool_kernel(
    const float* __restrict__ y, const float* __restrict__ w, const float* __restrict__ bias,
    ushort_t* __restrict__ outn, int Cc)
{
    int bc = blockIdx.x;                 // B*Cc
    int c = bc % Cc, b = bc / Cc;
    const float* p = y + (size_t)bc * 1024;

    float s = 0.f, ss = 0.f;
    for (int i = threadIdx.x; i < 1024; i += 256) { float v = p[i]; s += v; ss += v * v; }
    __shared__ float r1[256], r2[256];
    r1[threadIdx.x] = s; r2[threadIdx.x] = ss; __syncthreads();
    for (int st = 128; st > 0; st >>= 1) {
        if (threadIdx.x < st) { r1[threadIdx.x] += r1[threadIdx.x + st]; r2[threadIdx.x] += r2[threadIdx.x + st]; }
        __syncthreads();
    }
    float mean = r1[0] * (1.0f / 1024.0f);
    float var  = r2[0] * (1.0f / 1024.0f) - mean * mean;
    float rstd = rsqrtf(var + 1e-5f);
    float sw = w[c] * rstd;
    float sb = bias[c] - mean * sw;

    int head = c & 3, ch = c >> 2;
    if (LAY == 0) {
        ushort_t* o = outn + ((size_t)(b * 4 + head) * 1024) * 64 + ch;
        for (int j = threadIdx.x; j < 1024; j += 256)
            o[(size_t)j * 64] = f2bf(p[j] * sw + sb);
    } else {
        ushort_t* o = outn + ((size_t)(b * 4 + head) * 32 + ch) * 1024;
        for (int j = threadIdx.x; j < 1024; j += 256)
            o[j] = f2bf(p[j] * sw + sb);
    }
}

// ---------------------------------------------------------------------------
// Fused flash attention per (b, head, 64-row i-tile).
// 4 waves; each wave owns 16 query rows; online softmax over 16 j-blocks of 64.
// theta_n [bh, 4096, 64], phi_n [bh, 1024, 64], g_n [bh, 32, 1024],
// Obf out [b, i, 128] bf16 with channel = chv*4 + h.
// ---------------------------------------------------------------------------

__global__ __launch_bounds__(128) void attn_kernel(
    const ushort_t* __restrict__ thN, const ushort_t* __restrict__ phN,
    const ushort_t* __restrict__ gN, ushort_t* __restrict__ Obf)
{
    const int NB  = 4096 / 64;
    const int bid = blockIdx.x;
    const int ib  = bid % NB;
    const int bh  = bid / NB;
    const int h   = bh & 3;
    const int b   = bh >> 2;
    const int tid  = threadIdx.x;
    const int wave = tid >> 5;
    const int lane = tid & 31;
    const int ln   = lane & 15;
    const int hl   = lane >> 4;
    const int i0   = ib * 64 + wave * 16;

    __shared__ __align__(16) ushort_t Pl[4][16][64];   // per-wave P tile (bf16)

    // Resident theta A-fragments (M = query row, K = ch 0..63 -> 2 k-steps)
    FragB A[2];
    const ushort_t* tp = thN + ((size_t)bh * 4096 + i0 + ln) * 64;
#pragma unroll
    for (int ks = 0; ks < 2; ++ks) {
        A[ks].q[0] = *(const uint4*)(tp + ks * 32 + hl * 8);
        A[ks].q[1] = *(const uint4*)(tp + ks * 32 + 16 + hl * 8);
    }

    f32x8 Oc[2] = { zero8(), zero8() };
    float mr[8], lr[8];
#pragma unroll
    for (int r = 0; r < 8; ++r) { mr[r] = -1e30f; lr[r] = 0.0f; }

    const ushort_t* pbase = phN + (size_t)bh * 1024 * 64;
    const ushort_t* gbase = gN + (size_t)bh * 32 * 1024;

    for (int jb = 0; jb < 16; ++jb) {
        if (jb + 1 < 16) __builtin_prefetch(pbase + (size_t)(jb + 1) * 64 * 64, 0, 1);

        // ---- S tile: 16 rows x 64 cols (4 N-subtiles x 2 k-steps) ----
        f32x8 Sc[4];
#pragma unroll
        for (int jt = 0; jt < 4; ++jt) {
            Sc[jt] = zero8();
            const ushort_t* pp = pbase + (size_t)(jb * 64 + jt * 16 + ln) * 64;
#pragma unroll
            for (int ks = 0; ks < 2; ++ks) {
                FragB Bp;
                Bp.q[0] = *(const uint4*)(pp + ks * 32 + hl * 8);
                Bp.q[1] = *(const uint4*)(pp + ks * 32 + 16 + hl * 8);
                Sc[jt] = __builtin_amdgcn_wmma_f32_16x16x32_bf16(
                    false, A[ks].v, false, Bp.v, (short)0, Sc[jt], false, false);
            }
        }

        // ---- online softmax (rows r + 8*hl; reduce across the 16 col-lanes) ----
#pragma unroll
        for (int r = 0; r < 8; ++r) {
            float bm = fmaxf(fmaxf(Sc[0][r], Sc[1][r]), fmaxf(Sc[2][r], Sc[3][r]));
#pragma unroll
            for (int m = 1; m < 16; m <<= 1) bm = fmaxf(bm, __shfl_xor(bm, m, 32));
            float mn = fmaxf(mr[r], bm);
            float sc = __expf(mr[r] - mn);
            float rs = 0.0f;
#pragma unroll
            for (int jt = 0; jt < 4; ++jt) {
                float p = __expf(Sc[jt][r] - mn);
                Sc[jt][r] = p;
                rs += p;
            }
#pragma unroll
            for (int m = 1; m < 16; m <<= 1) rs += __shfl_xor(rs, m, 32);
            lr[r] = lr[r] * sc + rs;
            mr[r] = mn;
            Oc[0][r] *= sc;
            Oc[1][r] *= sc;
        }

        // ---- LDS round trip: C-layout P -> A-fragment layout (bf16) ----
        __syncthreads();
#pragma unroll
        for (int jt = 0; jt < 4; ++jt)
#pragma unroll
            for (int r = 0; r < 8; ++r)
                Pl[wave][r + hl * 8][jt * 16 + ln] = f2bf(Sc[jt][r]);
        __syncthreads();

        FragB Pf[2];
        const ushort_t* plp = &Pl[wave][ln][0];
#pragma unroll
        for (int ks = 0; ks < 2; ++ks) {
            Pf[ks].q[0] = *(const uint4*)(plp + ks * 32 + hl * 8);
            Pf[ks].q[1] = *(const uint4*)(plp + ks * 32 + 16 + hl * 8);
        }

        // ---- O += P @ g^T  (2 chv subtiles x 2 k-steps) ----
#pragma unroll
        for (int ct = 0; ct < 2; ++ct) {
#pragma unroll
            for (int ks = 0; ks < 2; ++ks) {
                FragB Gf;
                const ushort_t* gp = gbase + (size_t)(ct * 16 + ln) * 1024 + jb * 64 + ks * 32;
                Gf.q[0] = *(const uint4*)(gp + hl * 8);
                Gf.q[1] = *(const uint4*)(gp + 16 + hl * 8);
                Oc[ct] = __builtin_amdgcn_wmma_f32_16x16x32_bf16(
                    false, Pf[ks].v, false, Gf.v, (short)0, Oc[ct], false, false);
            }
        }
    }

    // ---- epilogue: divide by row sums, scatter to [b, i, chv*4 + h] ----
#pragma unroll
    for (int ct = 0; ct < 2; ++ct)
#pragma unroll
        for (int r = 0; r < 8; ++r) {
            float v = Oc[ct][r] / lr[r];
            int ig  = i0 + r + hl * 8;
            int chv = ct * 16 + ln;
            Obf[((size_t)b * 4096 + ig) * 128 + chv * 4 + h] = f2bf(v);
        }
}

// ---------------------------------------------------------------------------
// Host-side orchestration
// ---------------------------------------------------------------------------

extern "C" void kernel_launch(void* const* d_in, const int* in_sizes, int n_in,
                              void* d_out, int out_size, void* d_ws, size_t ws_size,
                              hipStream_t stream) {
    (void)in_sizes; (void)n_in; (void)out_size; (void)ws_size;

    const float* x      = (const float*)d_in[0];
    const float* thW    = (const float*)d_in[1];
    const float* phW    = (const float*)d_in[2];
    const float* gW     = (const float*)d_in[3];
    const float* oW     = (const float*)d_in[4];
    const float* nt_w   = (const float*)d_in[5];
    const float* nt_b   = (const float*)d_in[6];
    const float* np_w   = (const float*)d_in[7];
    const float* np_b   = (const float*)d_in[8];
    const float* ng_w   = (const float*)d_in[9];
    const float* ng_b   = (const float*)d_in[10];
    const float* gamma  = (const float*)d_in[11];
    float* out          = (float*)d_out;

    char* ws = (char*)d_ws;
    size_t off = 0;
    auto alloc = [&](size_t bytes) { char* p = ws + off; off += (bytes + 255) & ~size_t(255); return p; };

    // workspace layout (with aliasing: normalized tensors reuse dead conv bufs)
    ushort_t* xT    = (ushort_t*)alloc((size_t)8 * 4096 * 256 * 2);   // 16.78 MB
    ushort_t* wTh   = (ushort_t*)alloc(256 * 256 * 2);
    ushort_t* wPh   = (ushort_t*)alloc(256 * 256 * 2);
    ushort_t* wG    = (ushort_t*)alloc(128 * 256 * 2);
    ushort_t* wO    = (ushort_t*)alloc(256 * 128 * 2);
    float*    convT = (float*)alloc((size_t)8 * 256 * 4096 * 4);      // 33.55 MB
    float*    convP = (float*)alloc((size_t)8 * 256 * 4096 * 4);      // 33.55 MB (reused)
    float*    convG = (float*)alloc((size_t)8 * 128 * 4096 * 4);      // 16.78 MB (reused)
    float*    poolP = (float*)alloc((size_t)8 * 256 * 1024 * 4);      //  8.39 MB
    float*    poolG = (float*)alloc((size_t)8 * 128 * 1024 * 4);      //  4.19 MB
    // aliased into convP (dead after pooling):
    ushort_t* thN = (ushort_t*)convP;                                  // 16.78 MB
    ushort_t* phN = (ushort_t*)((char*)convP + (size_t)8 * 4096 * 256 * 2); // 4.19 MB
    // aliased into convG (dead after pooling):
    ushort_t* gN  = (ushort_t*)convG;                                  // 2.10 MB
    ushort_t* Obf = (ushort_t*)((char*)convG + (size_t)8 * 1024 * 128 * 2); // 8.39 MB

    // 1) pack weights + transposed activations to bf16
    pack_bf16<<<(256 * 256 + 255) / 256, 256, 0, stream>>>(thW, wTh, 256 * 256);
    pack_bf16<<<(256 * 256 + 255) / 256, 256, 0, stream>>>(phW, wPh, 256 * 256);
    pack_bf16<<<(128 * 256 + 255) / 256, 256, 0, stream>>>(gW, wG, 128 * 256);
    pack_bf16<<<(256 * 128 + 255) / 256, 256, 0, stream>>>(oW, wO, 256 * 128);
    pack_xT<<<(8 * 256 * 4096) / 256, 256, 0, stream>>>(x, xT);

    // 2) 1x1 convs as bf16 WMMA GEMMs (N = 32768 pixels, K = 256)
    gemm_bf16_kernel<0><<<dim3(256, 2), 256, 0, stream>>>(wTh, xT, convT, nullptr, nullptr, 256, 256);
    gemm_bf16_kernel<0><<<dim3(256, 2), 256, 0, stream>>>(wPh, xT, convP, nullptr, nullptr, 256, 256);
    gemm_bf16_kernel<0><<<dim3(256, 1), 256, 0, stream>>>(wG, xT, convG, nullptr, nullptr, 128, 256);

    // 3) 2x2 max pool for phi / g
    pool2_kernel<<<(8 * 256 * 1024) / 256, 256, 0, stream>>>(convP, poolP, 256);
    pool2_kernel<<<(8 * 128 * 1024) / 256, 256, 0, stream>>>(convG, poolG, 128);

    // 4) instance norms -> WMMA-friendly bf16 layouts
    inorm_theta_kernel<<<8 * 256, 256, 0, stream>>>(convT, nt_w, nt_b, thN);
    inorm_pool_kernel<0><<<8 * 256, 256, 0, stream>>>(poolP, np_w, np_b, phN, 256);
    inorm_pool_kernel<1><<<8 * 128, 256, 0, stream>>>(poolG, ng_w, ng_b, gN, 128);

    // 5) fused flash attention per (b, head, 64-row tile)
    attn_kernel<<<8 * 4 * (4096 / 64), 128, 0, stream>>>(thN, phN, gN, Obf);

    // 6) output 1x1 conv (K=128) with fused  out = gamma*o + x
    gemm_bf16_kernel<1><<<dim3(256, 2), 256, 0, stream>>>(wO, Obf, out, x, gamma, 256, 128);
}
